// MAXPrototypeContrastLoss_83846351553247
// MI455X (gfx1250) — matrix operand order repaired and checked by reading, not compiled, not run on hardware
//
#include <hip/hip_runtime.h>
#include <hip/hip_bf16.h>
#include <stdint.h>

// ---------------------------------------------------------------------------
// Problem constants (from reference): B=4, C=256, H=W=64, HW=4096, L=512
// ---------------------------------------------------------------------------
#define BB   4
#define CC   256
#define HH   64
#define WW   64
#define HWN  4096
#define LL   512

typedef __attribute__((ext_vector_type(16))) __bf16 v16bf;
typedef __attribute__((ext_vector_type(8)))  float  v8f;
typedef __attribute__((ext_vector_type(4)))  int    v4i;

union BfFrag {            // 16 bf16 = 32 bytes = two uint4
    uint4 u[2];
    v16bf v;
};

__device__ __forceinline__ unsigned short f2bf_rne(float f) {
    unsigned int u = __float_as_uint(f);
    unsigned int r = u + 0x7FFFu + ((u >> 16) & 1u);
    return (unsigned short)(r >> 16);
}

#if __has_builtin(__builtin_amdgcn_global_load_async_to_lds_b128)
#define MPCL_ASYNC_G2L 1
#else
#define MPCL_ASYNC_G2L 0
#endif

#if MPCL_ASYNC_G2L
typedef __attribute__((address_space(1))) v4i* g_v4i_ptr;   // global int4*
typedef __attribute__((address_space(3))) v4i* l_v4i_ptr;   // LDS int4*

// Async DMA: global (as1) -> LDS (as3), tracked by ASYNCcnt.
// Low 32 bits of a generic pointer to LDS are the LDS byte offset
// (flat shared aperture keeps the offset in ADDR[31:0]).
__device__ __forceinline__ void async_g2l_b128(const void* g, const void* lds) {
    __builtin_amdgcn_global_load_async_to_lds_b128(
        (g_v4i_ptr)(uintptr_t)g,
        (l_v4i_ptr)(unsigned)(uintptr_t)lds,
        0, 0);
}
__device__ __forceinline__ void wait_async0() {
#if __has_builtin(__builtin_amdgcn_s_wait_asynccnt)
    __builtin_amdgcn_s_wait_asynccnt(0);
#else
    asm volatile("s_wait_asynccnt 0" ::: "memory");
#endif
}
#endif

// ---------------------------------------------------------------------------
// Kernel 1: fused prep.
// One thread per (b, hw):
//   - nearest-resize labels (512 -> 64, stride 8)
//   - argmax of the 2-channel bg outputs, disrupt masks relu(1 - a - l)
//   - write negatives [B,2,HW,C] into d_out+1 (masked transpose of feats)
//   - stage pos = Sf*Sl as bf16 [b][s][c]  (A operand, c-contiguous)
//   - stage base^T = Qf*Ql as bf16 [b][q][c] (B operand, c-contiguous)
//   - column norms ||pos_s||, ||base_q||
// ---------------------------------------------------------------------------
__global__ void __launch_bounds__(256)
mpcl_prep_kernel(const float* __restrict__ Qf, const float* __restrict__ Sf,
                 const float* __restrict__ Qlab, const float* __restrict__ Slab,
                 const float* __restrict__ qbg, const float* __restrict__ sbg,
                 float* __restrict__ out /* [0]=loss, +1 negatives */,
                 unsigned short* __restrict__ Astage,
                 unsigned short* __restrict__ Bstage,
                 float* __restrict__ norm_pos,
                 float* __restrict__ norm_base) {
    const int tid = blockIdx.x * blockDim.x + threadIdx.x;   // 0 .. B*HW-1
    if (tid == 0) out[0] = 0.0f;                              // loss accumulator
    const int b  = tid / HWN;
    const int hw = tid % HWN;
    const int h  = hw >> 6;
    const int w  = hw & 63;

    // nearest resize: label[b, h*8, w*8]
    const float ql = Qlab[(size_t)b * LL * LL + (size_t)(h * 8) * LL + (w * 8)];
    const float sl = Slab[(size_t)b * LL * LL + (size_t)(h * 8) * LL + (w * 8)];

    // argmax over 2 channels (ties -> index 0, matching jnp.argmax)
    const float aq = (qbg[((size_t)(b * 2 + 1)) * HWN + hw] >
                      qbg[((size_t)(b * 2 + 0)) * HWN + hw]) ? 1.0f : 0.0f;
    const float as = (sbg[((size_t)(b * 2 + 1)) * HWN + hw] >
                      sbg[((size_t)(b * 2 + 0)) * HWN + hw]) ? 1.0f : 0.0f;

    const float qdis = fmaxf(1.0f - aq - ql, 0.0f);
    const float sdis = fmaxf(1.0f - as - sl, 0.0f);

    float* negQ = out + 1 + ((size_t)(b * 2 + 0) * HWN + hw) * CC;
    float* negS = out + 1 + ((size_t)(b * 2 + 1) * HWN + hw) * CC;
    unsigned int* arow = (unsigned int*)(Astage + ((size_t)b * HWN + hw) * CC);
    unsigned int* brow = (unsigned int*)(Bstage + ((size_t)b * HWN + hw) * CC);

    float sum_base = 0.0f, sum_pos = 0.0f;
    #pragma unroll 4
    for (int c = 0; c < CC; c += 2) {
        const float q0 = Qf[((size_t)(b * CC + c + 0)) * HWN + hw];
        const float q1 = Qf[((size_t)(b * CC + c + 1)) * HWN + hw];
        const float s0 = Sf[((size_t)(b * CC + c + 0)) * HWN + hw];
        const float s1 = Sf[((size_t)(b * CC + c + 1)) * HWN + hw];

        negQ[c + 0] = q0 * qdis;  negQ[c + 1] = q1 * qdis;
        negS[c + 0] = s0 * sdis;  negS[c + 1] = s1 * sdis;

        const float b0 = q0 * ql, b1 = q1 * ql;     // base column q=hw
        const float p0 = s0 * sl, p1 = s1 * sl;     // pos row s=hw
        sum_base += b0 * b0 + b1 * b1;
        sum_pos  += p0 * p0 + p1 * p1;

        brow[c >> 1] = (unsigned int)f2bf_rne(b0) |
                       ((unsigned int)f2bf_rne(b1) << 16);
        arow[c >> 1] = (unsigned int)f2bf_rne(p0) |
                       ((unsigned int)f2bf_rne(p1) << 16);
    }
    norm_base[(size_t)b * HWN + hw] = sqrtf(sum_base);
    norm_pos [(size_t)b * HWN + hw] = sqrtf(sum_pos);
}

// ---------------------------------------------------------------------------
// Kernel 2: WMMA bf16 GEMM with streaming max reduction over s.
// Block = 128 threads = 4 waves; each wave owns one 16-wide q-tile.
// - Double-buffered LDS A-tiles. Preferred fill path: async global->LDS DMA
//   (GLOBAL_LOAD_ASYNC_TO_LDS_B128, ASYNCcnt) issued for tile st+1 before the
//   WMMA chain of tile st; one s_wait_asynccnt 0 + barrier per iteration.
//   Fallback: branch-free register prefetch pinned by sched_barrier.
// - Two independent accumulator chains (even/odd K-steps) so two WMMAs are
//   always in flight per wave; ds loads for one chain overlap the other.
// - Fast v_rcp for the (ns*nq + eps) divide (well-conditioned, feeds a max).
// LDS rows padded to 264 bf16 (528B stride) => conflict-free ds_load_b128.
// D layout: VGPR r, lanes 0-15: M=r,N=lane ; lanes 16-31: M=r+8,N=lane-16.
// ---------------------------------------------------------------------------
__global__ void __launch_bounds__(128)
mpcl_gemm_kernel(const unsigned short* __restrict__ Astage,
                 const unsigned short* __restrict__ Bstage,
                 const float* __restrict__ norm_pos,
                 const float* __restrict__ norm_base,
                 float* __restrict__ simbuf) {
    __shared__ unsigned short As[2][16][264];   // 2 x (16 rows x 256 bf16 + pad)
    __shared__ float nsh[2][16];

    const int b     = blockIdx.x >> 6;            // 64 q-blocks per batch
    const int qblk  = blockIdx.x & 63;            // 64 q per block
    const int wave  = threadIdx.x >> 5;
    const int lane  = threadIdx.x & 31;
    const int qbase = qblk * 64 + wave * 16;
    const int lhalf = lane >> 4;                  // 0 or 1
    const int lmod  = lane & 15;

    // Per-thread A-tile staging slot: row = t/8, seg = t%8 (32 bf16 = 64B).
    const int srow = threadIdx.x >> 3;
    const int sseg = threadIdx.x & 7;
    const unsigned short* agbase =
        Astage + ((size_t)b * HWN + srow) * CC + sseg * 32;
    const float* npbase = norm_pos + (size_t)b * HWN + srow;

    // Preload this wave's B fragments (held across the whole s loop).
    // Lane l holds col N = l%16, K = (l/16)*8 + {0..7,16..23} for each K-step.
    BfFrag fb[8];
    {
        const unsigned short* bq =
            Bstage + ((size_t)b * HWN + qbase + lmod) * CC;
        #pragma unroll
        for (int kk = 0; kk < 8; ++kk) {
            const uint4* p = (const uint4*)(bq + kk * 32 + lhalf * 8);
            fb[kk].u[0] = p[0];
            fb[kk].u[1] = p[2];
        }
    }
    const float nq = norm_base[(size_t)b * HWN + qbase + lmod];

    constexpr int NS_TILES = HWN / 16;

    // Prologue: stage s-tile 0 into buffer 0. (8-way duplicate same-value
    // norm store is branch-free and benign.)
#if MPCL_ASYNC_G2L
    {
        const uint4* src = (const uint4*)agbase;
        #pragma unroll
        for (int i = 0; i < 4; ++i)
            async_g2l_b128(src + i, &As[0][srow][sseg * 32 + i * 8]);
        nsh[0][srow] = npbase[0];
        wait_async0();
    }
#else
    {
        const uint4* src = (const uint4*)agbase;
        uint4* dst = (uint4*)&As[0][srow][sseg * 32];
        #pragma unroll
        for (int i = 0; i < 4; ++i) dst[i] = src[i];
        nsh[0][srow] = npbase[0];
    }
#endif
    __syncthreads();

    float runmax = -3.4e38f;

    for (int st = 0; st < NS_TILES; ++st) {
        const int cur = st & 1;
        const int nxt = cur ^ 1;
        const int stn = (st + 1) & (NS_TILES - 1);   // wrap: branch-free

#if MPCL_ASYNC_G2L
        // Kick the DMA for the next tile into the other buffer now; it
        // overlaps the whole WMMA chain below.
        {
            const uint4* src = (const uint4*)(agbase + (size_t)stn * 16 * CC);
            #pragma unroll
            for (int i = 0; i < 4; ++i)
                async_g2l_b128(src + i, &As[nxt][srow][sseg * 32 + i * 8]);
        }
        const float pfn = npbase[(size_t)stn * 16];
#else
        // Register prefetch of the next tile, pinned before the WMMA chain.
        const uint4* src = (const uint4*)(agbase + (size_t)stn * 16 * CC);
        const uint4 pf0 = src[0], pf1 = src[1], pf2 = src[2], pf3 = src[3];
        const float pfn = npbase[(size_t)stn * 16];
#if __has_builtin(__builtin_amdgcn_sched_barrier)
        __builtin_amdgcn_sched_barrier(0);   // don't sink the loads
#endif
#endif

        // Two independent WMMA chains over K = 256 (8 steps of 32).
        v8f acc0, acc1;
        #pragma unroll
        for (int i = 0; i < 8; ++i) { acc0[i] = 0.0f; acc1[i] = 0.0f; }

        #pragma unroll
        for (int kk = 0; kk < 8; kk += 2) {
            // A layout: lanes 0-15 row M=l, K = k0+{0..7,16..23};
            //           lanes 16-31 row M=l-16, K = k0+{8..15,24..31}
            const int c0 = kk * 32 + lhalf * 8;
            BfFrag fa0, fa1;
            fa0.u[0] = *(const uint4*)&As[cur][lmod][c0];
            fa0.u[1] = *(const uint4*)&As[cur][lmod][c0 + 16];
            fa1.u[0] = *(const uint4*)&As[cur][lmod][c0 + 32];
            fa1.u[1] = *(const uint4*)&As[cur][lmod][c0 + 48];
            acc0 = __builtin_amdgcn_wmma_f32_16x16x32_bf16(
                false, fa0.v, false, fb[kk].v, (short)0, acc0, false, false);
            acc1 = __builtin_amdgcn_wmma_f32_16x16x32_bf16(
                false, fa1.v, false, fb[kk + 1].v, (short)0, acc1, false, false);
        }

        // Pos norms for this tile: lane needs rows lhalf*8 .. lhalf*8+7.
        const float4 n0 = *(const float4*)&nsh[cur][lhalf * 8];
        const float4 n1 = *(const float4*)&nsh[cur][lhalf * 8 + 4];
        const float nsv[8] = { n0.x, n0.y, n0.z, n0.w, n1.x, n1.y, n1.z, n1.w };

        // sim = num * rcp(ns*nq + eps); fold max over this s-tile.
        #pragma unroll
        for (int r = 0; r < 8; ++r) {
            const float num = acc0[r] + acc1[r];
            const float val = num * __builtin_amdgcn_rcpf(nsv[r] * nq + 1e-7f);
            runmax = fmaxf(runmax, val);
        }

        // Commit the next tile's norms (and, in fallback, the tile data).
#if MPCL_ASYNC_G2L
        nsh[nxt][srow] = pfn;
        wait_async0();          // this wave's DMA into As[nxt] has landed
#else
        {
            uint4* dst = (uint4*)&As[nxt][srow][sseg * 32];
            dst[0] = pf0; dst[1] = pf1; dst[2] = pf2; dst[3] = pf3;
            nsh[nxt][srow] = pfn;
        }
#endif
        __syncthreads();        // all waves' tiles landed / all readers done
    }

    // Max across the two lane halves (same N), lanes 0-15 write results.
    runmax = fmaxf(runmax, __shfl_xor(runmax, 16));
    if (lane < 16)
        simbuf[(size_t)b * HWN + qbase + lane] = runmax;
}

// ---------------------------------------------------------------------------
// Kernel 3: masked mean over fg query positions + log loss (one block per b).
// ---------------------------------------------------------------------------
__global__ void __launch_bounds__(256)
mpcl_reduce_kernel(const float* __restrict__ simbuf,
                   const float* __restrict__ Qlab,
                   float* __restrict__ out) {
    __shared__ float ssum[256];
    __shared__ float scnt[256];
    const int b = blockIdx.x;
    const int t = threadIdx.x;

    float sum = 0.0f, cnt = 0.0f;
    for (int q = t; q < HWN; q += 256) {
        const int h = q >> 6, w = q & 63;
        const float l = Qlab[(size_t)b * LL * LL + (size_t)(h * 8) * LL + (w * 8)];
        const float m = (l == 1.0f) ? 1.0f : 0.0f;
        sum += simbuf[(size_t)b * HWN + q] * m;
        cnt += m;
    }
    ssum[t] = sum;
    scnt[t] = cnt;
    __syncthreads();
    for (int s = 128; s > 0; s >>= 1) {
        if (t < s) { ssum[t] += ssum[t + s]; scnt[t] += scnt[t + s]; }
        __syncthreads();
    }
    if (t == 0) {
        const float pc = 0.5f * (ssum[0] / fmaxf(scnt[0], 1.0f) + 1.0f);
        atomicAdd(out, -logf(pc + 1e-8f) * (1.0f / (float)BB));
    }
}

// ---------------------------------------------------------------------------
// Launch
// ---------------------------------------------------------------------------
extern "C" void kernel_launch(void* const* d_in, const int* in_sizes, int n_in,
                              void* d_out, int out_size, void* d_ws, size_t ws_size,
                              hipStream_t stream) {
    (void)in_sizes; (void)n_in; (void)out_size; (void)ws_size;
    const float* Qf   = (const float*)d_in[0];
    const float* Sf   = (const float*)d_in[1];
    // d_in[2] = Q_predit (unused by reference)
    const float* Qlab = (const float*)d_in[3];
    const float* Slab = (const float*)d_in[4];
    const float* qbg  = (const float*)d_in[5];
    const float* sbg  = (const float*)d_in[6];
    // d_in[7] = classes (unused by reference)
    float* out = (float*)d_out;

    // Workspace carve-up (16 MiB staging + 192 KiB norms/sim)
    unsigned short* Astage = (unsigned short*)d_ws;                    // pos bf16 [B][s][c]
    unsigned short* Bstage = Astage + (size_t)BB * HWN * CC;           // base^T bf16 [B][q][c]
    float* norm_pos  = (float*)(Bstage + (size_t)BB * HWN * CC);
    float* norm_base = norm_pos + (size_t)BB * HWN;
    float* simbuf    = norm_base + (size_t)BB * HWN;

    mpcl_prep_kernel<<<(BB * HWN) / 256, 256, 0, stream>>>(
        Qf, Sf, Qlab, Slab, qbg, sbg, out, Astage, Bstage, norm_pos, norm_base);

    mpcl_gemm_kernel<<<BB * (HWN / 64), 128, 0, stream>>>(
        Astage, Bstage, norm_pos, norm_base, simbuf);

    mpcl_reduce_kernel<<<BB, 256, 0, stream>>>(simbuf, Qlab, out);
}
